// MultiheadAttention_54468775248215
// MI455X (gfx1250) — compile-verified, matrix operand order checked by reading, not run on hardware
//
#include <hip/hip_runtime.h>

// MI455X (gfx1250) multi-head attention pipeline, bf16 WMMA with f32 accum.
// Workspace (bf16): wbf 4x256x256 weights (0.5 MB), q/k [b][h][s][d] (8 MB ea),
// vT [b][h][d][s] (8 MB), attn [b][s][c] (8 MB).

#define CH   256
#define NH   8
#define HD   32
#define SEQ  1024
#define NB   16
#define XSTR 264   // padded LDS row stride (bf16 elems), 528 B = 16B-aligned
#define PSTR 40    // padded LDS row stride for P tile, 80 B = 16B-aligned

typedef __attribute__((ext_vector_type(16))) __bf16 v16bf;
typedef __attribute__((ext_vector_type(8)))  __bf16 v8bf;
typedef __attribute__((ext_vector_type(8)))  float  v8f;

// Native fptrunc: backend emits hardware bf16 convert (packed when paired).
static __device__ __forceinline__ __bf16 f2bf(float f) { return (__bf16)f; }
static __device__ __forceinline__ unsigned pack2bf(float a, float b) {
    union { __bf16 h[2]; unsigned u; } c;
    c.h[0] = (__bf16)a; c.h[1] = (__bf16)b;
    return c.u;
}

// 16x32 bf16 operand fragment (A-layout; B mirrors with lane = N).
// Lane's 16 elements = two contiguous 8-elem runs -> two 16-byte loads.
static __device__ __forceinline__ v16bf ld_frag(const __bf16* row, int half) {
    const v8bf lo = *(const v8bf*)(row + (half << 3));
    const v8bf hi = *(const v8bf*)(row + 16 + (half << 3));
    v16bf r;
#pragma unroll
    for (int j = 0; j < 8; ++j) { r[j] = lo[j]; r[j + 8] = hi[j]; }
    return r;
}

// 16x16x256 tile matmul: A-frags preloaded, B-frags software-pipelined so the
// k-step kt+1 loads are in flight while the kt WMMA executes.
static __device__ __forceinline__ v8f tile_mm256(const __bf16* arow,
                                                 const __bf16* brow, int half) {
    v16bf aa[8];
#pragma unroll
    for (int kt = 0; kt < 8; ++kt) aa[kt] = ld_frag(arow + kt * 32, half);
    v8f d = {0.f, 0.f, 0.f, 0.f, 0.f, 0.f, 0.f, 0.f};
    v16bf bcur = ld_frag(brow, half);
#pragma unroll
    for (int kt = 0; kt < 8; ++kt) {
        v16bf bnext = bcur;
        if (kt < 7) bnext = ld_frag(brow + (kt + 1) * 32, half);
        d = __builtin_amdgcn_wmma_f32_16x16x32_bf16(
                false, aa[kt], false, bcur, (short)0, d, false, false);
        bcur = bnext;
    }
    return d;
}

// ---- CDNA5 async LDS staging (ASYNCcnt path) ------------------------------
static __device__ __forceinline__ void async_cp16(unsigned lds_addr, const void* gaddr) {
    asm volatile("global_load_async_to_lds_b128 %0, %1, off"
                 :: "v"(lds_addr), "v"(gaddr) : "memory");
}
static __device__ __forceinline__ void wait_async0() {
    asm volatile("s_wait_asynccnt 0x0" ::: "memory");
}
static __device__ __forceinline__ unsigned lds_off(const void* p) {
    return (unsigned)(size_t)p;        // LDS aperture: addr[31:0] = LDS offset
}

// ---------------------------------------------------------------------------
// Kernel 0: one-shot fp32 -> bf16 conversion of the four 256x256 weights.
// ---------------------------------------------------------------------------
__global__ __launch_bounds__(256) void wcvt_kernel(
    const float* __restrict__ Wq, const float* __restrict__ Wk,
    const float* __restrict__ Wv, const float* __restrict__ Wo,
    __bf16* __restrict__ dst)
{
    typedef __attribute__((ext_vector_type(4))) __bf16 v4bf;
    int t  = blockIdx.x * 256 + threadIdx.x;      // 0 .. 65535
    int mi = t >> 14;                             // matrix id (uniform per block)
    int off = (t & 16383) << 2;
    const float* src = (mi == 0) ? Wq : (mi == 1) ? Wk : (mi == 2) ? Wv : Wo;
    const float4 f = *(const float4*)(src + off);
    v4bf o; o[0] = f2bf(f.x); o[1] = f2bf(f.y); o[2] = f2bf(f.z); o[3] = f2bf(f.w);
    *(v4bf*)(dst + mi * 65536 + off) = o;
}

// ---------------------------------------------------------------------------
// Kernel 1: Q/K/V projections.  grid = NB * (SEQ/64), block = 256 (8 waves).
// Three which-literal tile loops: EXEC stays all-ones, no predicated stores.
// ---------------------------------------------------------------------------
__global__ __launch_bounds__(256) void qkv_proj_kernel(
    const float* __restrict__ x, const __bf16* __restrict__ wbf,
    __bf16* __restrict__ qw, __bf16* __restrict__ kw, __bf16* __restrict__ vtw)
{
    __shared__ __align__(16) __bf16 xt[64 * XSTR];   // [64 tokens][256 ch]
    const int b  = blockIdx.x >> 4;
    const int s0 = (blockIdx.x & 15) << 6;
    const int tid = threadIdx.x;

    unsigned* xtu = (unsigned*)xt;
#pragma unroll 4
    for (int t = tid; t < 64 * (CH / 2); t += 256) {
        int si = t & 63, c2 = t >> 6;
        const float* px = x + ((size_t)b * CH + c2 * 2) * SEQ + s0 + si;
        xtu[si * (XSTR / 2) + c2] = pack2bf(px[0], px[SEQ]);
    }
    __syncthreads();

    const int wave = tid >> 5, lane = tid & 31, half = lane >> 4, ln = lane & 15;

    // --- Q tiles (strided [s][d] stores) ---
    for (int T = wave; T < 64; T += 8) {
        int m0 = (T & 3) << 4, nb0 = (T >> 2) << 4;
        v8f d = tile_mm256(xt + (m0 + ln) * XSTR,
                           wbf + (nb0 + ln) * CH, half);
        int dout = nb0 + ln, hh = dout >> 5, dd = dout & 31;
#pragma unroll
        for (int i = 0; i < 8; ++i) {
            int s = s0 + m0 + (half << 3) + i;
            qw[(((size_t)(b * NH + hh) * SEQ) + s) * HD + dd] = f2bf(d[i]);
        }
    }
    // --- K tiles ---
    for (int T = wave; T < 64; T += 8) {
        int m0 = (T & 3) << 4, nb0 = (T >> 2) << 4;
        v8f d = tile_mm256(xt + (m0 + ln) * XSTR,
                           wbf + 65536 + (nb0 + ln) * CH, half);
        int dout = nb0 + ln, hh = dout >> 5, dd = dout & 31;
#pragma unroll
        for (int i = 0; i < 8; ++i) {
            int s = s0 + m0 + (half << 3) + i;
            kw[(((size_t)(b * NH + hh) * SEQ) + s) * HD + dd] = f2bf(d[i]);
        }
    }
    // --- V tiles (transposed [d][s] layout -> contiguous 16B stores) ---
    for (int T = wave; T < 64; T += 8) {
        int m0 = (T & 3) << 4, nb0 = (T >> 2) << 4;
        v8f d = tile_mm256(xt + (m0 + ln) * XSTR,
                           wbf + 2 * 65536 + (nb0 + ln) * CH, half);
        int dout = nb0 + ln, hh = dout >> 5, dd = dout & 31;
        v8bf vv;
#pragma unroll
        for (int i = 0; i < 8; ++i) vv[i] = f2bf(d[i]);
        *(v8bf*)(vtw + (((size_t)(b * NH + hh) * HD) + dd) * SEQ
                     + s0 + m0 + (half << 3)) = vv;
    }
}

// ---------------------------------------------------------------------------
// Kernel 2: flash attention with double-buffered async K/V staging.
// grid = NB*NH*(SEQ/64), block = 128 (4 waves); each wave owns 16 query rows.
// ---------------------------------------------------------------------------
__global__ __launch_bounds__(128) void attn_kernel(
    const __bf16* __restrict__ qw, const __bf16* __restrict__ kw,
    const __bf16* __restrict__ vtw, __bf16* __restrict__ aw)
{
    __shared__ __align__(16) __bf16 kbuf[2][32 * 32];   // [kv][d]
    __shared__ __align__(16) __bf16 vbuf[2][32 * 32];   // [d][kv]
    __shared__ __align__(16) __bf16 lds_p[4][16 * PSTR];

    const int tid = threadIdx.x;
    const int wave = tid >> 5, lane = tid & 31, half = lane >> 4, ln = lane & 15;
    const int qt = blockIdx.x & 15;
    const int h  = (blockIdx.x >> 4) & 7;
    const int b  = blockIdx.x >> 7;
    const int bh = b * NH + h;
    const int q0 = (qt << 6) + (wave << 4);

    const __bf16* krows = kw + (size_t)bh * SEQ * HD;
    const __bf16* vrows = vtw + (size_t)bh * HD * SEQ;

    // flat 16B-per-thread copy mapping: row = tid/4, col = (tid%4)*8 elems
    const int crow = tid >> 2;
    const int ccol = (tid & 3) << 3;
    const unsigned klds[2] = { lds_off(&kbuf[0][tid * 8]), lds_off(&kbuf[1][tid * 8]) };
    const unsigned vlds[2] = { lds_off(&vbuf[0][tid * 8]), lds_off(&vbuf[1][tid * 8]) };

    v16bf qA = ld_frag(qw + ((size_t)bh * SEQ + q0 + ln) * HD, half);

    float mrow[8], lrow[8];
    v8f acc0 = {0.f, 0.f, 0.f, 0.f, 0.f, 0.f, 0.f, 0.f};
    v8f acc1 = {0.f, 0.f, 0.f, 0.f, 0.f, 0.f, 0.f, 0.f};
#pragma unroll
    for (int i = 0; i < 8; ++i) { mrow[i] = -3.0e38f; lrow[i] = 0.f; }

    const float scale = 0.17677669529663687f;   // 1/sqrt(32)
    const float L2E   = 1.4426950408889634f;
    __bf16* pt = &lds_p[wave][0];
    const v8f zero = {0.f, 0.f, 0.f, 0.f, 0.f, 0.f, 0.f, 0.f};

    // prologue: stage chunk 0
    async_cp16(klds[0], krows + (size_t)crow * HD + ccol);
    async_cp16(vlds[0], vrows + (size_t)crow * SEQ + ccol);

    for (int c = 0; c < SEQ / 32; ++c) {
        const int kv0 = c << 5;
        wait_async0();          // my chunk-c portions have landed in LDS
        __syncthreads();        // everyone's chunk-c landed; chunk c-1 reads done

        if (c + 1 < SEQ / 32) { // overlap next chunk's DMA with this compute
            const int nb = (c + 1) & 1;
            async_cp16(klds[nb], krows + (size_t)(kv0 + 32 + crow) * HD + ccol);
            async_cp16(vlds[nb], vrows + (size_t)crow * SEQ + kv0 + 32 + ccol);
        }

        const __bf16* kt_ = &kbuf[c & 1][0];
        const __bf16* vt_ = &vbuf[c & 1][0];

        v16bf kB0 = ld_frag(kt_ + ln * HD, half);
        v16bf kB1 = ld_frag(kt_ + (16 + ln) * HD, half);
        v8f sc0 = __builtin_amdgcn_wmma_f32_16x16x32_bf16(
                      false, qA, false, kB0, (short)0, zero, false, false);
        v8f sc1 = __builtin_amdgcn_wmma_f32_16x16x32_bf16(
                      false, qA, false, kB1, (short)0, zero, false, false);

#pragma unroll
        for (int i = 0; i < 8; ++i) {        // row M = i + 8*half over 16 lanes
            float a = sc0[i] * scale, c2 = sc1[i] * scale;
            float mx = fmaxf(a, c2);
            mx = fmaxf(mx, __shfl_xor(mx, 1, 32));
            mx = fmaxf(mx, __shfl_xor(mx, 2, 32));
            mx = fmaxf(mx, __shfl_xor(mx, 4, 32));
            mx = fmaxf(mx, __shfl_xor(mx, 8, 32));
            float mnew = fmaxf(mrow[i], mx);
            float corr = __builtin_exp2f((mrow[i] - mnew) * L2E);
            mrow[i] = mnew;
            float p0 = __builtin_exp2f((a  - mnew) * L2E);
            float p1 = __builtin_exp2f((c2 - mnew) * L2E);
            float rs = p0 + p1;
            rs += __shfl_xor(rs, 1, 32);
            rs += __shfl_xor(rs, 2, 32);
            rs += __shfl_xor(rs, 4, 32);
            rs += __shfl_xor(rs, 8, 32);
            lrow[i] = lrow[i] * corr + rs;
            acc0[i] *= corr;
            acc1[i] *= corr;
            int r = (half << 3) + i;
            pt[r * PSTR + ln]      = f2bf(p0);
            pt[r * PSTR + 16 + ln] = f2bf(p1);
        }
        // same-wave LDS RAW on the P tile
        asm volatile("s_wait_dscnt 0x0" ::: "memory");

        v16bf pA  = ld_frag(pt + ln * PSTR, half);
        v16bf vB0 = ld_frag(vt_ + ln * 32, half);
        v16bf vB1 = ld_frag(vt_ + (16 + ln) * 32, half);
        acc0 = __builtin_amdgcn_wmma_f32_16x16x32_bf16(
                   false, pA, false, vB0, (short)0, acc0, false, false);
        acc1 = __builtin_amdgcn_wmma_f32_16x16x32_bf16(
                   false, pA, false, vB1, (short)0, acc1, false, false);
    }

    // normalize and store to [b][s][c] with c = h*32 + d
#pragma unroll
    for (int i = 0; i < 8; ++i) {
        float inv = 1.0f / lrow[i];
        int s = q0 + (half << 3) + i;
        size_t base = ((size_t)b * SEQ + s) * CH + h * HD;
        aw[base + ln]      = f2bf(acc0[i] * inv);
        aw[base + 16 + ln] = f2bf(acc1[i] * inv);
    }
}

// ---------------------------------------------------------------------------
// Kernel 3: output projection, z[b][dout][s] = sum_c attn[b][s][c] * Wo[dout][c]
// ---------------------------------------------------------------------------
__global__ __launch_bounds__(256) void out_proj_kernel(
    const __bf16* __restrict__ aw, const __bf16* __restrict__ wo_bf,
    float* __restrict__ out)
{
    const int b  = blockIdx.x >> 4;
    const int s0 = (blockIdx.x & 15) << 6;
    const int tid = threadIdx.x;
    const int wave = tid >> 5, lane = tid & 31, half = lane >> 4, ln = lane & 15;

    for (int T = wave; T < 64; T += 8) {
        int m0 = (T & 3) << 4, n0 = (T >> 2) << 4;
        v8f d = tile_mm256(aw + ((size_t)b * SEQ + s0 + m0 + ln) * CH,
                           wo_bf + (n0 + ln) * CH, half);
        int dout = n0 + ln;
        float* o = out + ((size_t)b * CH + dout) * SEQ + s0 + m0 + (half << 3);
        *(float4*)(o)     = make_float4(d[0], d[1], d[2], d[3]);
        *(float4*)(o + 4) = make_float4(d[4], d[5], d[6], d[7]);
    }
}

// ---------------------------------------------------------------------------
extern "C" void kernel_launch(void* const* d_in, const int* in_sizes, int n_in,
                              void* d_out, int out_size, void* d_ws, size_t ws_size,
                              hipStream_t stream) {
    const float* x  = (const float*)d_in[0];
    const float* Wq = (const float*)d_in[1];
    const float* Wk = (const float*)d_in[2];
    const float* Wv = (const float*)d_in[3];
    const float* Wo = (const float*)d_in[4];
    float* out = (float*)d_out;

    const size_t qkv_elems = (size_t)NB * NH * SEQ * HD;   // 4 Mi each
    __bf16* wbf = (__bf16*)d_ws;                           // 4 x 64 Ki
    __bf16* qw  = wbf + 4 * 65536;
    __bf16* kw  = qw  + qkv_elems;
    __bf16* vtw = kw  + qkv_elems;
    __bf16* aw  = vtw + qkv_elems;                         // total ~32.5 MB

    wcvt_kernel<<<256, 256, 0, stream>>>(Wq, Wk, Wv, Wo, wbf);
    qkv_proj_kernel<<<NB * (SEQ / 64), 256, 0, stream>>>(x, wbf, qw, kw, vtw);
    attn_kernel<<<NB * NH * (SEQ / 64), 128, 0, stream>>>(qw, kw, vtw, aw);
    out_proj_kernel<<<NB * (SEQ / 64), 256, 0, stream>>>(aw, Wo == nullptr ? wbf : wbf + 3 * 65536, out);
}
